// DirectMPIGO_46119358824510
// MI455X (gfx1250) — matrix operand order repaired and checked by reading, not compiled, not run on hardware
//
#include <hip/hip_runtime.h>
#include <hip/hip_bf16.h>

// K-planes-style fused kernel for MI455X (gfx1250, wave32).
// Phase 1: multi-scale tri-plane bilinear gather -> 64 f16 features/point in LDS.
// Phase 2: fused 64->128->128->64 MLP via v_wmma_f32_16x16x32_f16, per-wave
//          16-point M tiles (the whole pipeline is wave-local by construction).

typedef __attribute__((ext_vector_type(16))) _Float16 v16h;
typedef __attribute__((ext_vector_type(8)))  float    v8f;

union Frag16 { v16h h; unsigned int u[8]; };

// A-matrix fragment (16x32 f16, M x K), source row-major [M][K] halves.
// ISA layout: lane M = lane%16; dword j holds K pair at
//   K = k0 + 2j + 8*[j>=4] + 8*[lane>=16]
__device__ __forceinline__ v16h load_a_frag(const _Float16* base, int row,
                                            int strideHalves, int k0, int lane) {
  Frag16 f;
  const unsigned int* p = (const unsigned int*)(base + row * strideHalves);
  const int hi = (lane >> 4) & 1;
#pragma unroll
  for (int j = 0; j < 8; ++j) {
    const int k = k0 + 2 * j + ((j >= 4) ? 8 : 0) + 8 * hi;
    f.u[j] = p[k >> 1];
  }
  return f.h;
}

// B-matrix fragment (32x16 f16, K x N) from row-major weights [N][K] halves
// (B = W^T). ISA layout: lane N = lane%16; dword j holds K pair at
//   K = k0 + 2j + 8*[j>=4] + 8*[lane>=16]
__device__ __forceinline__ v16h load_b_frag(const _Float16* w, int nrow,
                                            int strideHalves, int k0, int lane) {
  Frag16 f;
  const unsigned int* p = (const unsigned int*)(w + nrow * strideHalves);
  const int kb = (k0 + 8 * ((lane >> 4) & 1)) >> 1;
#pragma unroll
  for (int j = 0; j < 8; ++j) f.u[j] = p[kb + j + ((j >= 4) ? 4 : 0)];
  return f.h;
}

__global__ void convert_weights_kernel(const float* __restrict__ w0,
                                       const float* __restrict__ w1,
                                       const float* __restrict__ w2,
                                       _Float16* __restrict__ wsh) {
  const int i = blockIdx.x * blockDim.x + threadIdx.x;
  if (i < 128 * 64)                     wsh[i] = (_Float16)w0[i];
  else if (i < 128 * 64 + 128 * 128)    wsh[i] = (_Float16)w1[i - 128 * 64];
  else if (i < 128 * 64 + 128 * 128 + 64 * 128)
                                        wsh[i] = (_Float16)w2[i - 128 * 64 - 128 * 128];
}

__global__ __launch_bounds__(256)
void kplanes_fused_kernel(const float* __restrict__ pts,
                          const float* __restrict__ p0, const float* __restrict__ p1,
                          const float* __restrict__ p2, const float* __restrict__ p3,
                          const _Float16* __restrict__ w0h,
                          const _Float16* __restrict__ w1h,
                          const _Float16* __restrict__ w2h,
                          const float* __restrict__ b0, const float* __restrict__ b1,
                          const float* __restrict__ b2,
                          float* __restrict__ out, int N) {
  // smemA: feats (first 128x64 halves) then reused as h2 (128x128). smemB: h1.
  __shared__ __align__(16) _Float16 smemA[128 * 128];
  __shared__ __align__(16) _Float16 smemB[128 * 128];
  _Float16* feats = smemA;
  _Float16* h1    = smemB;
  _Float16* h2    = smemA;

  const int tid   = threadIdx.x;
  const int baseP = blockIdx.x * 128;

  // ---------- Phase 1: multi-scale tri-plane bilinear features ----------
  {
    const int pl = tid >> 1;      // local point 0..127
    const int hp = tid & 1;       // which scale pair this thread computes
    int pg = baseP + pl;
    if (pg >= N) pg = N - 1;
    const float4 pt = ((const float4*)pts)[pg];
    const float coord[3] = {pt.x, pt.y, pt.z};

    const int H = 150;
    float iy = (pt.w + 1.0f) * 0.5f * (float)(H - 1);
    iy = fminf(fmaxf(iy, 0.0f), (float)(H - 1));
    const int   iy0 = (int)iy;
    const int   iy1 = min(iy0 + 1, H - 1);
    const float wy  = iy - (float)iy0;

#pragma unroll
    for (int si = 0; si < 2; ++si) {
      const int s = hp * 2 + si;
      const float* plane = (s == 0) ? p0 : (s == 1) ? p1 : (s == 2) ? p2 : p3;
      const int W  = 64 << s;
      const int HW = H * W;

      float acc[16];
#pragma unroll
      for (int c = 0; c < 16; ++c) acc[c] = 1.0f;

#pragma unroll
      for (int j = 0; j < 3; ++j) {
        float ix = (coord[j] + 1.0f) * 0.5f * (float)(W - 1);
        ix = fminf(fmaxf(ix, 0.0f), (float)(W - 1));
        const int   ix0 = (int)ix;
        const int   ix1 = min(ix0 + 1, W - 1);
        const float wx  = ix - (float)ix0;

        const float c00 = (1.0f - wx) * (1.0f - wy);
        const float c01 = wx * (1.0f - wy);
        const float c10 = (1.0f - wx) * wy;
        const float c11 = wx * wy;
        const int o00 = iy0 * W + ix0, o01 = iy0 * W + ix1;
        const int o10 = iy1 * W + ix0, o11 = iy1 * W + ix1;
        const float* pb = plane + j * 16 * HW;
#pragma unroll
        for (int c = 0; c < 16; ++c) {
          const float* cb = pb + c * HW;
          acc[c] *= cb[o00] * c00 + cb[o01] * c01 + cb[o10] * c10 + cb[o11] * c11;
        }
      }
#pragma unroll
      for (int c = 0; c < 16; ++c)
        feats[pl * 64 + s * 16 + c] = (_Float16)acc[c];
    }
  }
  __syncthreads();

  // ---------- Phase 2: fused MLP with WMMA ----------
  const int lane = tid & 31;
  const int wave = tid >> 5;            // M-tile index (16 points per wave)
  const int nlo  = lane & 15;
  const int hi8  = ((lane >> 4) & 1) * 8;
  const int mrow = wave * 16 + nlo;

  // Layer 0: feats[128x64] @ w0^T -> relu -> h1[128x128]
  {
    const v16h a0 = load_a_frag(feats, mrow, 64, 0,  lane);
    const v16h a1 = load_a_frag(feats, mrow, 64, 32, lane);
#pragma unroll
    for (int nt = 0; nt < 8; ++nt) {
      const int n = nt * 16 + nlo;
      v8f c = {};
      const v16h bf0 = load_b_frag(w0h, n, 64, 0, lane);
      c = __builtin_amdgcn_wmma_f32_16x16x32_f16(false, a0, false, bf0, (short)0, c, false, false);
      const v16h bf1 = load_b_frag(w0h, n, 64, 32, lane);
      c = __builtin_amdgcn_wmma_f32_16x16x32_f16(false, a1, false, bf1, (short)0, c, false, false);
      const float bias = b0[n];
#pragma unroll
      for (int r = 0; r < 8; ++r) {
        float v = c[r] + bias;
        v = v > 0.0f ? v : 0.0f;
        h1[(wave * 16 + r + hi8) * 128 + n] = (_Float16)v;
      }
    }
  }
  __syncthreads();

  // Layer 1: h1[128x128] @ w1^T -> relu -> h2[128x128]
  {
    v16h a[4];
#pragma unroll
    for (int ks = 0; ks < 4; ++ks) a[ks] = load_a_frag(h1, mrow, 128, ks * 32, lane);
#pragma unroll
    for (int nt = 0; nt < 8; ++nt) {
      const int n = nt * 16 + nlo;
      v8f c = {};
#pragma unroll
      for (int ks = 0; ks < 4; ++ks) {
        const v16h bf = load_b_frag(w1h, n, 128, ks * 32, lane);
        c = __builtin_amdgcn_wmma_f32_16x16x32_f16(false, a[ks], false, bf, (short)0, c, false, false);
      }
      const float bias = b1[n];
#pragma unroll
      for (int r = 0; r < 8; ++r) {
        float v = c[r] + bias;
        v = v > 0.0f ? v : 0.0f;
        h2[(wave * 16 + r + hi8) * 128 + n] = (_Float16)v;
      }
    }
  }
  __syncthreads();

  // Layer 2: h2[128x128] @ w2^T + b2 -> out[128x64] (f32, no relu)
  {
    v16h a[4];
#pragma unroll
    for (int ks = 0; ks < 4; ++ks) a[ks] = load_a_frag(h2, mrow, 128, ks * 32, lane);
#pragma unroll
    for (int nt = 0; nt < 4; ++nt) {
      const int n = nt * 16 + nlo;
      v8f c = {};
#pragma unroll
      for (int ks = 0; ks < 4; ++ks) {
        const v16h bf = load_b_frag(w2h, n, 128, ks * 32, lane);
        c = __builtin_amdgcn_wmma_f32_16x16x32_f16(false, a[ks], false, bf, (short)0, c, false, false);
      }
      const float bias = b2[n];
#pragma unroll
      for (int r = 0; r < 8; ++r) {
        const int pg = baseP + wave * 16 + r + hi8;
        if (pg < N) out[pg * 64 + n] = c[r] + bias;
      }
    }
  }
}

extern "C" void kernel_launch(void* const* d_in, const int* in_sizes, int n_in,
                              void* d_out, int out_size, void* d_ws, size_t ws_size,
                              hipStream_t stream) {
  const float* pts = (const float*)d_in[0];
  const float* p0  = (const float*)d_in[1];
  const float* p1  = (const float*)d_in[2];
  const float* p2  = (const float*)d_in[3];
  const float* p3  = (const float*)d_in[4];
  const float* w0  = (const float*)d_in[5];
  const float* b0  = (const float*)d_in[6];
  const float* w1  = (const float*)d_in[7];
  const float* b1  = (const float*)d_in[8];
  const float* w2  = (const float*)d_in[9];
  const float* b2  = (const float*)d_in[10];
  float* out = (float*)d_out;

  const int N = in_sizes[0] / 4;

  // f16 weight copies in workspace: w0h[128x64] | w1h[128x128] | w2h[64x128]
  _Float16* wsh = (_Float16*)d_ws;
  convert_weights_kernel<<<128, 256, 0, stream>>>(w0, w1, w2, wsh);
  const _Float16* w0h = wsh;
  const _Float16* w1h = wsh + 128 * 64;
  const _Float16* w2h = wsh + 128 * 64 + 128 * 128;

  const int blocks = (N + 127) / 128;
  kplanes_fused_kernel<<<blocks, 256, 0, stream>>>(pts, p0, p1, p2, p3,
                                                   w0h, w1h, w2h, b0, b1, b2,
                                                   out, N);
}
